// FineGrainedRetriever_5609227289000
// MI455X (gfx1250) — compile-verified, matrix operand order checked by reading, not run on hardware
//
#include <hip/hip_runtime.h>
#include <math.h>
#include <stdint.h>

#define E_EDGES 200000
#define NN_NODES 50000
#define EMB 256

typedef __attribute__((ext_vector_type(16))) __bf16 v16bf;
typedef __attribute__((ext_vector_type(8)))  float  v8f;

struct GemmArgs {
  // MODE 0: A = [A0 | A1] concat along K (A1 may be null, K0==K)
  const float* A0; int lda0; int K0;
  const float* A1; int lda1;
  // MODE 1: fused gather A = [q, h_e[h], edge_attr, h_e[t]], K = 1536
  const float* q; const float* e0; const float* e1; const float* eattr;
  const int* hsrc; const int* tdst;
  // common
  const float* W;     // [K, N] row-major
  const float* bias;  // [N]
  float* C; int ldc;
  int M, N, K, relu;
};

__device__ __forceinline__ uint32_t pkbf(float lo, float hi) {
  __bf16 l = (__bf16)lo, h = (__bf16)hi;
  uint32_t lu = (uint32_t)__builtin_bit_cast(unsigned short, l);
  uint32_t hu = (uint32_t)__builtin_bit_cast(unsigned short, h);
  return lu | (hu << 16);
}

// C[M,N] = act(A[M,K] @ W[K,N] + bias). 128x128 tile per block, BK=32,
// double-buffered LDS with register-staged global loads, plus a two-deep
// register pipeline on B fragments so ds_load latency hides under WMMA.
// 8 waves; wave w computes rows [w*16, w*16+16) x 128 cols (8 accum frags).
template <int MODE>
__global__ __launch_bounds__(256) void gemm_bf16_wmma(GemmArgs g) {
  // packed bf16 pairs; row stride 20 u32 = 80B (16B-aligned, bank-padded)
  __shared__ uint32_t AsB[2][128][20];  // [buf][row][kpair]
  __shared__ uint32_t BsB[2][128][20];  // [buf][col][kpair] (transposed W)

  const int tid  = threadIdx.x;
  const int wave = tid >> 5;
  const int lane = tid & 31;
  const int m0 = blockIdx.x * 128;
  const int n0 = blockIdx.y * 128;

  v8f acc[8];
  #pragma unroll
  for (int j = 0; j < 8; ++j)
    #pragma unroll
    for (int r = 0; r < 8; ++r) acc[j][r] = 0.0f;

  const int arow = tid >> 1;         // 0..127 : A tile row this thread stages
  const int ak16 = (tid & 1) << 4;   // 0 / 16 : which 16 k's
  const int bkp  = tid >> 4;         // 0..15  : k-pair (2 k rows of W)
  const int bn8  = (tid & 15) << 3;  // 0..120 : 8 consecutive n's

  uint32_t aReg[8], bReg[8];

  auto loadA = [&](int kt) {
    const int row = m0 + arow;
    const int k0  = kt + ak16;
    const float* p = nullptr;
    if (row < g.M) {
      if (MODE == 0) {
        p = (k0 < g.K0) ? g.A0 + (size_t)row * g.lda0 + k0
                        : g.A1 + (size_t)row * g.lda1 + (k0 - g.K0);
      } else {
        // pieces: [0,256)=q  [256,768)=h_e[h]  [768,1024)=edge_attr  [1024,1536)=h_e[t]
        if (k0 < EMB) {
          p = g.q + k0;
        } else if (k0 < 3 * EMB) {
          int c = k0 - EMB;
          int n = g.hsrc[row];
          p = (c < EMB) ? g.e0 + (size_t)n * EMB + c
                        : g.e1 + (size_t)n * EMB + (c - EMB);
        } else if (k0 < 4 * EMB) {
          p = g.eattr + (size_t)row * EMB + (k0 - 3 * EMB);
        } else {
          int c = k0 - 4 * EMB;
          int n = g.tdst[row];
          p = (c < EMB) ? g.e0 + (size_t)n * EMB + c
                        : g.e1 + (size_t)n * EMB + (c - EMB);
        }
      }
    }
    if (p) {
      #pragma unroll
      for (int v = 0; v < 4; ++v) {
        float4 f = ((const float4*)p)[v];   // global_load_b128
        aReg[2 * v]     = pkbf(f.x, f.y);
        aReg[2 * v + 1] = pkbf(f.z, f.w);
      }
    } else {
      #pragma unroll
      for (int i = 0; i < 8; ++i) aReg[i] = 0u;
    }
  };

  auto loadB = [&](int kt) {
    const float* r0 = g.W + (size_t)(kt + 2 * bkp)     * g.N + (n0 + bn8);
    const float* r1 = g.W + (size_t)(kt + 2 * bkp + 1) * g.N + (n0 + bn8);
    float4 a0 = ((const float4*)r0)[0];
    float4 a1 = ((const float4*)r0)[1];
    float4 c0 = ((const float4*)r1)[0];
    float4 c1 = ((const float4*)r1)[1];
    bReg[0] = pkbf(a0.x, c0.x); bReg[1] = pkbf(a0.y, c0.y);
    bReg[2] = pkbf(a0.z, c0.z); bReg[3] = pkbf(a0.w, c0.w);
    bReg[4] = pkbf(a1.x, c1.x); bReg[5] = pkbf(a1.y, c1.y);
    bReg[6] = pkbf(a1.z, c1.z); bReg[7] = pkbf(a1.w, c1.w);
  };

  auto stage = [&](int buf) {
    // A: two ds_store_b128 (16B-aligned: row*80 + {0,32})
    uint4 u0 = {aReg[0], aReg[1], aReg[2], aReg[3]};
    uint4 u1 = {aReg[4], aReg[5], aReg[6], aReg[7]};
    *(uint4*)&AsB[buf][arow][(ak16 >> 1) + 0] = u0;
    *(uint4*)&AsB[buf][arow][(ak16 >> 1) + 4] = u1;
    // B transposed: eight b32 (merged to ds_store_2addr_b32 by the compiler)
    #pragma unroll
    for (int i = 0; i < 8; ++i) BsB[buf][bn8 + i][bkp] = bReg[i];
  };

  // 16-bit A/B layout (ISA 7.12.2): lane l -> row/col = l&15; lanes 0-15
  // hold K = 0..7,16..23 ; lanes 16-31 hold K = 8..15,24..31.
  const int kb = (lane >> 4) << 3;

  auto loadBfrag = [&](int buf, int j, v16bf& b) {
    const __bf16* bp = (const __bf16*)&BsB[buf][(j << 4) + (lane & 15)][0];
    #pragma unroll
    for (int i = 0; i < 8; ++i) {
      b[i]     = bp[kb + i];
      b[8 + i] = bp[kb + 16 + i];
    }
  };

  auto compute = [&](int buf) {
    const int mr = (wave << 4) + (lane & 15);
    const __bf16* ap = (const __bf16*)&AsB[buf][mr][0];
    v16bf a;
    #pragma unroll
    for (int i = 0; i < 8; ++i) {
      a[i]     = ap[kb + i];
      a[8 + i] = ap[kb + 16 + i];
    }
    // two-deep B-fragment pipeline: issue loads for j+1, WMMA consumes j,
    // so the wait before each WMMA is dscnt<=2 instead of 0.
    v16bf bp0, bp1;
    loadBfrag(buf, 0, bp0);
    #pragma unroll
    for (int j = 0; j < 8; ++j) {
      if (j < 7) loadBfrag(buf, j + 1, (j & 1) ? bp0 : bp1);
      acc[j] = __builtin_amdgcn_wmma_f32_16x16x32_bf16(
          false, a, false, (j & 1) ? bp1 : bp0, (short)0, acc[j], false, false);
    }
  };

  // software pipeline: regs hold tile kt+32 while LDS[buf] feeds WMMA for kt
  loadA(0);
  loadB(0);
  int buf = 0;
  for (int kt = 0; kt < g.K; kt += 32) {
    stage(buf);
    __syncthreads();
    if (kt + 32 < g.K) {
      loadA(kt + 32);
      loadB(kt + 32);
    }
    compute(buf);
    buf ^= 1;
  }

  // ---- epilogue: bias + relu + store ----
  // C/D f32 layout: VGPR r, lane l -> M = r + (l>>4)*8, N = l&15.
  #pragma unroll
  for (int j = 0; j < 8; ++j) {
    const int col = n0 + (j << 4) + (lane & 15);
    const float bv = g.bias ? g.bias[col] : 0.0f;
    #pragma unroll
    for (int r = 0; r < 8; ++r) {
      const int row = m0 + (wave << 4) + r + ((lane >> 4) << 3);
      if (row < g.M) {
        float v = acc[j][r] + bv;
        if (g.relu) v = fmaxf(v, 0.0f);
        g.C[(size_t)row * g.ldc + col] = v;
      }
    }
  }
}

__global__ void zero_f32(float* p, size_t n) {
  size_t i = (size_t)blockIdx.x * blockDim.x + threadIdx.x;
  if (i < n) p[i] = 0.0f;
}

__global__ void count_deg(const int* hsrc, const int* tdst, float* cnt) {
  int e = blockIdx.x * blockDim.x + threadIdx.x;
  if (e < E_EDGES) {
    atomicAdd(&cnt[tdst[e]], 1.0f);
    atomicAdd(&cnt[hsrc[e]], 1.0f);
  }
}

// Both directions of the bidirectional edge list:
//   fwd: msg = x[h] + edge_attr[e] -> agg[t]
//   rev: msg = x[t] + ea_rev[e]    -> agg[h]
__global__ __launch_bounds__(256) void sage_scatter(
    const float* x, const float* eattr, const float* earev,
    const int* hsrc, const int* tdst, float* agg) {
  const int e = blockIdx.x;
  const int c = threadIdx.x;
  const int h = hsrc[e];
  const int t = tdst[e];
  float v0 = x[(size_t)h * EMB + c] + eattr[(size_t)e * EMB + c];
  atomicAdd(&agg[(size_t)t * EMB + c], v0);
  float v1 = x[(size_t)t * EMB + c] + earev[(size_t)e * EMB + c];
  atomicAdd(&agg[(size_t)h * EMB + c], v1);
}

__global__ __launch_bounds__(256) void div_cnt(float* agg, const float* cnt) {
  const int n = blockIdx.x;
  const int c = threadIdx.x;
  agg[(size_t)n * EMB + c] /= fmaxf(cnt[n], 1.0f);
}

// Wave-per-edge GEMV (256-dot) + bias + sigmoid(logit + logit(noise)).
__global__ __launch_bounds__(256) void final_pred(
    const float* h1, const float* w2, const float* b2,
    const float* noise, float* out) {
  const int wave = threadIdx.x >> 5;
  const int lane = threadIdx.x & 31;
  const int e = blockIdx.x * 8 + wave;
  if (e >= E_EDGES) return;
  const float* row = h1 + (size_t)e * EMB;
  float s = 0.0f;
  #pragma unroll
  for (int i = 0; i < EMB / 32; ++i) {
    int c = lane + i * 32;
    s += row[c] * w2[c];
  }
  #pragma unroll
  for (int m = 16; m > 0; m >>= 1) s += __shfl_xor(s, m, 32);
  if (lane == 0) {
    float logit = s + b2[0];
    out[e] = logit;
    float nz = noise[e];
    float rn = logf(nz) - logf(1.0f - nz);
    out[E_EDGES + e] = 1.0f / (1.0f + expf(-(logit + rn)));
  }
}

extern "C" void kernel_launch(void* const* d_in, const int* in_sizes, int n_in,
                              void* d_out, int out_size, void* d_ws, size_t ws_size,
                              hipStream_t stream) {
  const float* entity = (const float*)d_in[0];
  const float* eattr  = (const float*)d_in[1];
  const float* q      = (const float*)d_in[2];
  const float* noise  = (const float*)d_in[3];
  const float* W_pr1  = (const float*)d_in[4];
  const float* b_pr1  = (const float*)d_in[5];
  const float* W_pr2  = (const float*)d_in[6];
  const float* b_pr2  = (const float*)d_in[7];
  const float* sW0    = (const float*)d_in[8];
  const float* sb0    = (const float*)d_in[9];
  const float* sW1    = (const float*)d_in[10];
  const float* sb1    = (const float*)d_in[11];
  const float* pW1    = (const float*)d_in[12];
  const float* pb1    = (const float*)d_in[13];
  const float* pW2    = (const float*)d_in[14];
  const float* pb2    = (const float*)d_in[15];
  const int*   ei     = (const int*)d_in[16];
  const int* hsrc = ei;             // edge_index[0]
  const int* tdst = ei + E_EDGES;   // edge_index[1]

  float* ws = (float*)d_ws;
  size_t off = 0;
  float* earev = ws + off; off += (size_t)E_EDGES * EMB;   // 51.2M
  float* buf   = ws + off; off += (size_t)E_EDGES * EMB;   // tmp1, later h1
  float* out0  = ws + off; off += (size_t)NN_NODES * EMB;
  float* out1  = ws + off; off += (size_t)NN_NODES * EMB;
  float* agg   = ws + off; off += (size_t)NN_NODES * EMB;
  float* cnt   = ws + off; off += (size_t)NN_NODES;
  (void)ws_size; (void)in_sizes; (void)n_in; (void)out_size;

  dim3 blk(256);
  auto grid_gemm = [](int M, int N) {
    return dim3((unsigned)((M + 127) / 128), (unsigned)(N / 128));
  };

  // 1) tmp1 = relu(edge_attr @ W_pr1 + b_pr1)
  {
    GemmArgs g{};
    g.A0 = eattr; g.lda0 = EMB; g.K0 = EMB;
    g.W = W_pr1; g.bias = b_pr1; g.C = buf; g.ldc = EMB;
    g.M = E_EDGES; g.N = EMB; g.K = EMB; g.relu = 1;
    gemm_bf16_wmma<0><<<grid_gemm(E_EDGES, EMB), blk, 0, stream>>>(g);
  }
  // 2) ea_rev = tmp1 @ W_pr2 + b_pr2
  {
    GemmArgs g{};
    g.A0 = buf; g.lda0 = EMB; g.K0 = EMB;
    g.W = W_pr2; g.bias = b_pr2; g.C = earev; g.ldc = EMB;
    g.M = E_EDGES; g.N = EMB; g.K = EMB; g.relu = 0;
    gemm_bf16_wmma<0><<<grid_gemm(E_EDGES, EMB), blk, 0, stream>>>(g);
  }
  // 3) degree counts (same for both layers)
  zero_f32<<<(NN_NODES + 255) / 256, blk, 0, stream>>>(cnt, (size_t)NN_NODES);
  count_deg<<<(E_EDGES + 255) / 256, blk, 0, stream>>>(hsrc, tdst, cnt);

  // 4) SAGE layers
  const float* x = entity;
  float* outs[2] = {out0, out1};
  const float* Ws[2] = {sW0, sW1};
  const float* bs[2] = {sb0, sb1};
  for (int l = 0; l < 2; ++l) {
    size_t an = (size_t)NN_NODES * EMB;
    zero_f32<<<(unsigned)((an + 255) / 256), blk, 0, stream>>>(agg, an);
    sage_scatter<<<E_EDGES, blk, 0, stream>>>(x, eattr, earev, hsrc, tdst, agg);
    div_cnt<<<NN_NODES, blk, 0, stream>>>(agg, cnt);
    GemmArgs g{};
    g.A0 = x;   g.lda0 = EMB; g.K0 = EMB;   // concat [x, agg] without materializing
    g.A1 = agg; g.lda1 = EMB;
    g.W = Ws[l]; g.bias = bs[l]; g.C = outs[l]; g.ldc = EMB;
    g.M = NN_NODES; g.N = EMB; g.K = 2 * EMB; g.relu = 1;
    gemm_bf16_wmma<0><<<grid_gemm(NN_NODES, EMB), blk, 0, stream>>>(g);
    x = outs[l];
  }

  // 5) pred layer 1 with fused gather: h1 = relu(h_triple @ pred_W1 + pred_b1)
  //    (h_e = [out0|out1] is ~102MB -> L2-resident gathers). Reuses `buf`.
  {
    GemmArgs g{};
    g.q = q; g.e0 = out0; g.e1 = out1; g.eattr = eattr;
    g.hsrc = hsrc; g.tdst = tdst;
    g.W = pW1; g.bias = pb1; g.C = buf; g.ldc = EMB;
    g.M = E_EDGES; g.N = EMB; g.K = 6 * EMB; g.relu = 1;
    gemm_bf16_wmma<1><<<grid_gemm(E_EDGES, EMB), blk, 0, stream>>>(g);
  }

  // 6) logits + idp-bern sampling -> d_out = [logits(E); sampled(E)]
  final_pred<<<(E_EDGES + 7) / 8, blk, 0, stream>>>(buf, pW2, pb2, noise, (float*)d_out);
}
